// GNNTower_70746701299879
// MI455X (gfx1250) — matrix operand order repaired and checked by reading, not compiled, not run on hardware
//
#include <hip/hip_runtime.h>
#include <hip/hip_bf16.h>

#define NN   100000
#define EE   1600000
#define IN_XF 20
#define HT   32
#define HX   128
#define HY   64
#define HCATF 608
#define OUTF 20
#define EPSF 1e-5f

typedef __attribute__((ext_vector_type(16))) __bf16 v16bf;
typedef __attribute__((ext_vector_type(8)))  float  v8f;

__device__ __forceinline__ v8f wmma_bf16(v16bf a, v16bf b, v8f c) {
  return __builtin_amdgcn_wmma_f32_16x16x32_bf16(false, a, false, b, (short)0, c,
                                                 false, false);
}

// A fragment: 16x32 (MxK) bf16, ISA layout. base pre-offset to row m0.
template <typename T>
__device__ __forceinline__ v16bf load_a(const T* base, int rowStride, int kValid,
                                        int k0, int lane) {
  int m = lane & 15, half = lane >> 4;
  v16bf a;
#pragma unroll
  for (int i = 0; i < 16; i++) {
    int vv = i >> 1, lo = i & 1;
    int k  = ((vv < 4) ? 0 : 16) + half * 8 + ((vv & 3) << 1) + lo;
    int kk = k0 + k;
    float x = (kk < kValid) ? (float)base[(size_t)m * rowStride + kk] : 0.f;
    a[i] = (__bf16)x;
  }
  return a;
}

// Packed-B fragment: 16 contiguous bf16 per lane (two b128 loads).
__device__ __forceinline__ v16bf load_b_pk(const __bf16* P, int S, int t, int s,
                                           int lane) {
  return *(const v16bf*)(P + ((((size_t)t * S + s) * 32 + lane) << 4));
}

// ---- weight pre-pack: fp32 row-major K x Nc -> bf16 fragment order ----
__global__ void k_pack(const float* W, int K, int Nc, int S, int T, __bf16* out) {
  int idx = blockIdx.x * 256 + threadIdx.x;   // over T*S*512
  if (idx >= T * S * 512) return;
  int i  = idx & 15;
  int l  = (idx >> 4) & 31;
  int st = idx >> 9;
  int s = st % S, t = st / S;
  int n = t * 16 + (l & 15);
  int k = s * 32 + ((i >> 3) << 4) + (l >> 4) * 8 + (i & 7);
  float x = (k < K && n < Nc) ? W[(size_t)k * Nc + n] : 0.f;
  out[idx] = (__bf16)x;
}

// ---- tiny t-MLP: (1)->(32)->(32) ----
__global__ void k_ht(const float* t, const float* W1, const float* b1,
                     const float* W2, const float* b2, float* ht) {
  __shared__ float h1[HT];
  int o = threadIdx.x;
  h1[o] = fmaxf(fmaf(t[0], W1[o], b1[o]), 0.f);
  __syncthreads();
  float acc = b2[o];
  for (int k = 0; k < HT; k++) acc = fmaf(h1[k], W2[k * HT + o], acc);
  ht[o] = fmaxf(acc, 0.f);
}

// ---- embedding gather + broadcast t into hcat slices ----
__global__ void k_embed(const int* Y, const float* emb, const float* ht, float* hcat) {
  int i = blockIdx.x * blockDim.x + threadIdx.x;   // over N*96
  int node = i / 96, f = i - node * 96;
  size_t row = (size_t)node * HCATF;
  if (f < HY) hcat[row + 384 + f] = emb[Y[node] * HY + f];
  else        hcat[row + 576 + (f - HY)] = ht[f - HY];
}

// ---- X MLP: (N,20)@W1 relu @W2 relu -> hcat X0 slice. 16 nodes/block, 8 waves. ----
__global__ void k_hx(const float* X, const __bf16* W1p, const float* b1,
                     const __bf16* W2p, const float* b2, float* hcat) {
  __shared__ float h1[16 * HX];
  int tid = threadIdx.x, lane = tid & 31, w = tid >> 5;
  int node0 = blockIdx.x * 16;
  int half = lane >> 4;

  v8f c = {};
  v16bf a = load_a(X + (size_t)node0 * IN_XF, IN_XF, IN_XF, 0, lane);
  v16bf b = load_b_pk(W1p, 1, w, 0, lane);
  c = wmma_bf16(a, b, c);
  {
    int n = w * 16 + (lane & 15);
    float bias = b1[n];
#pragma unroll
    for (int r = 0; r < 8; r++)
      h1[(r + half * 8) * HX + n] = fmaxf(c[r] + bias, 0.f);
  }
  __syncthreads();

  v8f c2 = {};
  for (int ks = 0; ks < 4; ks++) {
    v16bf a2  = load_a(h1, HX, HX, ks * 32, lane);
    v16bf b2v = load_b_pk(W2p, 4, w, ks, lane);
    c2 = wmma_bf16(a2, b2v, c2);
  }
  {
    int n = w * 16 + (lane & 15);
    float bias = b2[n];
#pragma unroll
    for (int r = 0; r < 8; r++)
      hcat[(size_t)(node0 + r + half * 8) * HCATF + n] = fmaxf(c2[r] + bias, 0.f);
  }
}

__global__ void k_zero(float* p, int n) {
  int i = blockIdx.x * blockDim.x + threadIdx.x;
  if (i < n) p[i] = 0.f;
}

// ---- edge scatter: aggr[dst] += [h_X|h_Y][src], width 192 ----
__global__ void k_scatter(const float* hcat, const int* src, const int* dst,
                          float* aggr, int xoff, int yoff) {
  int i = blockIdx.x * 256 + threadIdx.x;   // over E*192
  int e = i / 192, f = i - e * 192;
  int s = src[e], d = dst[e];
  float v = (f < HX) ? hcat[(size_t)s * HCATF + xoff + f]
                     : hcat[(size_t)s * HCATF + yoff + (f - HX)];
  atomicAdd(&aggr[(size_t)d * 192 + f], v);
}

// ---- layer X update: LN(relu([aggr(192)|ht(32)] @ uX.W + b)) ----
__global__ void k_updX(const float* aggr, const float* ht, const __bf16* Wp,
                       const float* ub, const float* g, const float* bb,
                       float* hcat, int outoff) {
  __shared__ float hs[16 * HX];
  int tid = threadIdx.x, lane = tid & 31, w = tid >> 5;
  int node0 = blockIdx.x * 16;
  int m = lane & 15, half = lane >> 4;

  v8f c = {};
  for (int ks = 0; ks < 7; ks++) {
    int k0 = ks * 32;
    v16bf a;
#pragma unroll
    for (int i = 0; i < 16; i++) {
      int vv = i >> 1, lo = i & 1;
      int k  = ((vv < 4) ? 0 : 16) + half * 8 + ((vv & 3) << 1) + lo;
      int kk = k0 + k;
      float x = (kk < 192) ? aggr[(size_t)(node0 + m) * 192 + kk] : ht[kk - 192];
      a[i] = (__bf16)x;
    }
    v16bf b = load_b_pk(Wp, 7, w, ks, lane);
    c = wmma_bf16(a, b, c);
  }
  {
    int n = w * 16 + (lane & 15);
    float bias = ub[n];
#pragma unroll
    for (int r = 0; r < 8; r++)
      hs[(r + half * 8) * HX + n] = fmaxf(c[r] + bias, 0.f);
  }
  __syncthreads();

  int node = tid >> 4, j = tid & 15;
  float s = 0.f, sq = 0.f;
#pragma unroll
  for (int q = 0; q < 8; q++) {
    float x = hs[node * HX + j + q * 16]; s += x; sq += x * x;
  }
  for (int msk = 1; msk < 16; msk <<= 1) {
    s  += __shfl_xor(s,  msk, 16);
    sq += __shfl_xor(sq, msk, 16);
  }
  float mean = s * (1.f / HX);
  float var  = sq * (1.f / HX) - mean * mean;
  float rstd = rsqrtf(var + EPSF);
#pragma unroll
  for (int q = 0; q < 8; q++) {
    int col = j + q * 16;
    float x = hs[node * HX + col];
    hcat[(size_t)(node0 + node) * HCATF + outoff + col] =
        (x - mean) * rstd * g[col] + bb[col];
  }
}

// ---- layer Y update: LN(relu(aggr_Y(64) @ uY.W + b)), 4 waves/block ----
__global__ void k_updY(const float* aggr, const __bf16* Wp, const float* ub,
                       const float* g, const float* bb, float* hcat, int outoff) {
  __shared__ float hs[16 * HY];
  int tid = threadIdx.x, lane = tid & 31, w = tid >> 5;
  int node0 = blockIdx.x * 16;
  int m = lane & 15, half = lane >> 4;

  v8f c = {};
  for (int ks = 0; ks < 2; ks++) {
    int k0 = ks * 32;
    v16bf a;
#pragma unroll
    for (int i = 0; i < 16; i++) {
      int vv = i >> 1, lo = i & 1;
      int k  = ((vv < 4) ? 0 : 16) + half * 8 + ((vv & 3) << 1) + lo;
      float x = aggr[(size_t)(node0 + m) * 192 + 128 + k0 + k];
      a[i] = (__bf16)x;
    }
    v16bf b = load_b_pk(Wp, 2, w, ks, lane);
    c = wmma_bf16(a, b, c);
  }
  {
    int n = w * 16 + (lane & 15);
    float bias = ub[n];
#pragma unroll
    for (int r = 0; r < 8; r++)
      hs[(r + half * 8) * HY + n] = fmaxf(c[r] + bias, 0.f);
  }
  __syncthreads();

  int node = tid >> 3, j = tid & 7;
  float s = 0.f, sq = 0.f;
#pragma unroll
  for (int q = 0; q < 8; q++) {
    float x = hs[node * HY + j + q * 8]; s += x; sq += x * x;
  }
  for (int msk = 1; msk < 8; msk <<= 1) {
    s  += __shfl_xor(s,  msk, 8);
    sq += __shfl_xor(sq, msk, 8);
  }
  float mean = s * (1.f / HY);
  float var  = sq * (1.f / HY) - mean * mean;
  float rstd = rsqrtf(var + EPSF);
#pragma unroll
  for (int q = 0; q < 8; q++) {
    int col = j + q * 8;
    float x = hs[node * HY + col];
    hcat[(size_t)(node0 + node) * HCATF + outoff + col] =
        (x - mean) * rstd * g[col] + bb[col];
  }
}

// ---- output head: relu(hcat @ o1) @ o2 -> logits ----
__global__ void k_out(const float* hcat, const __bf16* W1p, const float* b1,
                      const __bf16* W2p, const float* b2, float* out) {
  __shared__ float  aF[16 * HCATF];       // 38912 B
  __shared__ __bf16 hidLds[16 * HCATF];   // 19456 B
  int tid = threadIdx.x, lane = tid & 31, w = tid >> 5;
  int node0 = blockIdx.x * 16;
  int half = lane >> 4;

  // async-stage the 16x608 fp32 activation tile straight into LDS
  {
    unsigned ldsbase = (unsigned)(size_t)&aF[0];
    unsigned long long gbase =
        (unsigned long long)(size_t)(hcat + (size_t)node0 * HCATF);
    for (int i = tid; i < (16 * HCATF) / 2; i += 256) {   // 8-byte chunks
      unsigned off = (unsigned)(i * 8);
      asm volatile("global_load_async_to_lds_b64 %0, %1, %2"
                   :: "v"(ldsbase + off), "v"(off), "s"(gbase)
                   : "memory");
    }
    asm volatile("s_wait_asynccnt 0x0" ::: "memory");
  }
  __syncthreads();

  for (int tile = w; tile < 38; tile += 8) {
    v8f c = {};
    for (int ks = 0; ks < 19; ks++) {
      v16bf a = load_a(aF, HCATF, HCATF, ks * 32, lane);
      v16bf b = load_b_pk(W1p, 19, tile, ks, lane);
      c = wmma_bf16(a, b, c);
    }
    int n = tile * 16 + (lane & 15);
    float bias = b1[n];
#pragma unroll
    for (int r = 0; r < 8; r++)
      hidLds[(r + half * 8) * HCATF + n] = (__bf16)fmaxf(c[r] + bias, 0.f);
  }
  __syncthreads();

  if (w < 2) {
    v8f c = {};
    for (int ks = 0; ks < 19; ks++) {
      v16bf a = load_a(hidLds, HCATF, HCATF, ks * 32, lane);
      v16bf b = load_b_pk(W2p, 19, w, ks, lane);
      c = wmma_bf16(a, b, c);
    }
    int n = w * 16 + (lane & 15);
    if (n < OUTF) {
      float bias = b2[n];
#pragma unroll
      for (int r = 0; r < 8; r++)
        out[(size_t)(node0 + r + half * 8) * OUTF + n] = c[r] + bias;
    }
  }
}

extern "C" void kernel_launch(void* const* d_in, const int* in_sizes, int n_in,
                              void* d_out, int out_size, void* d_ws, size_t ws_size,
                              hipStream_t stream) {
  // params pytree flattened in sorted-key order
  const float* emb = (const float*)d_in[0];
  const float *LbX[2], *LbY[2], *LgX[2], *LgY[2], *LuXW[2], *LuXb[2], *LuYW[2], *LuYb[2];
  for (int l = 0; l < 2; l++) {
    int b = 1 + l * 8;
    LbX[l]  = (const float*)d_in[b + 0];
    LbY[l]  = (const float*)d_in[b + 1];
    LgX[l]  = (const float*)d_in[b + 2];
    LgY[l]  = (const float*)d_in[b + 3];
    LuXW[l] = (const float*)d_in[b + 4];
    LuXb[l] = (const float*)d_in[b + 5];
    LuYW[l] = (const float*)d_in[b + 6];
    LuYb[l] = (const float*)d_in[b + 7];
  }
  const float* o1W = (const float*)d_in[17]; const float* o1b = (const float*)d_in[18];
  const float* o2W = (const float*)d_in[19]; const float* o2b = (const float*)d_in[20];
  const float* t1W = (const float*)d_in[21]; const float* t1b = (const float*)d_in[22];
  const float* t2W = (const float*)d_in[23]; const float* t2b = (const float*)d_in[24];
  const float* x1W = (const float*)d_in[25]; const float* x1b = (const float*)d_in[26];
  const float* x2W = (const float*)d_in[27]; const float* x2b = (const float*)d_in[28];
  const float* t   = (const float*)d_in[29];
  const float* X   = (const float*)d_in[30];
  const int*   Y   = (const int*)d_in[31];
  const int*   src = (const int*)d_in[32];
  const int*   dst = (const int*)d_in[33];
  float* out = (float*)d_out;

  float* ws   = (float*)d_ws;
  float* hcat = ws;                                  // N*608 f32
  float* aggr = ws + (size_t)NN * HCATF;             // N*192 f32
  float* ht   = ws + (size_t)NN * (HCATF + 192);     // 32 f32

  // packed bf16 weights (32B-aligned: N*800*4 bytes is divisible by 32)
  __bf16* pk   = (__bf16*)(ws + (size_t)NN * (HCATF + 192) + 32);
  __bf16* x1p  = pk;               // 8*1*512   = 4096
  __bf16* x2p  = x1p + 4096;       // 8*4*512   = 16384
  __bf16* uXp0 = x2p + 16384;      // 8*7*512   = 28672
  __bf16* uXp1 = uXp0 + 28672;
  __bf16* uYp0 = uXp1 + 28672;     // 4*2*512   = 4096
  __bf16* uYp1 = uYp0 + 4096;
  __bf16* o1p  = uYp1 + 4096;      // 38*19*512 = 369664
  __bf16* o2p  = o1p + 369664;     // 2*19*512  = 19456
  __bf16* uXp[2] = {uXp0, uXp1};
  __bf16* uYp[2] = {uYp0, uYp1};

  auto packN = [&](const float* W, int K, int Nc, int S, int T, __bf16* outp) {
    int total = T * S * 512;
    k_pack<<<(total + 255) / 256, 256, 0, stream>>>(W, K, Nc, S, T, outp);
  };
  packN(x1W, 20, 128, 1, 8, x1p);
  packN(x2W, 128, 128, 4, 8, x2p);
  packN(LuXW[0], 224, 128, 7, 8, uXp0);
  packN(LuXW[1], 224, 128, 7, 8, uXp1);
  packN(LuYW[0], 64, 64, 2, 4, uYp0);
  packN(LuYW[1], 64, 64, 2, 4, uYp1);
  packN(o1W, 608, 608, 19, 38, o1p);
  packN(o2W, 608, 20, 19, 2, o2p);

  k_ht<<<1, 32, 0, stream>>>(t, t1W, t1b, t2W, t2b, ht);
  k_embed<<<(NN * 96) / 256, 256, 0, stream>>>(Y, emb, ht, hcat);
  k_hx<<<NN / 16, 256, 0, stream>>>(X, x1p, x1b, x2p, x2b, hcat);

  for (int l = 0; l < 2; l++) {
    int xoff = HX * l, yoff = 384 + HY * l;
    k_zero<<<(NN * 192) / 256, 256, 0, stream>>>(aggr, NN * 192);
    k_scatter<<<(EE / 256) * 192, 256, 0, stream>>>(hcat, src, dst, aggr, xoff, yoff);
    k_updX<<<NN / 16, 256, 0, stream>>>(aggr, ht, uXp[l], LuXb[l], LgX[l], LbX[l],
                                        hcat, HX * (l + 1));
    k_updY<<<NN / 16, 128, 0, stream>>>(aggr, uYp[l], LuYb[l], LgY[l], LbY[l],
                                        hcat, 384 + HY * (l + 1));
  }

  k_out<<<NN / 16, 256, 0, stream>>>(hcat, o1p, o1b, o2p, o2b, out);
}
